// WindowAttention_56676388438019
// MI455X (gfx1250) — matrix operand order repaired and checked by reading, not compiled, not run on hardware
//
#include <hip/hip_runtime.h>

typedef __attribute__((ext_vector_type(16))) _Float16 v16h;
typedef __attribute__((ext_vector_type(8)))  float    v8f;

#define N_TOK 64
#define DIM   256
#define HEADS 8
#define HD    32
#define SCALE 0.17677669529663689f   // 1/sqrt(32)

// LDS strides in _Float16 units (padded for conflict-free b128 access)
#define XS_STR 264   // 256 + 8   (132 dwords -> lane bank step 4  : conflict free)
#define QK_STR 40    // 32 + 8    (20 dwords  -> lane bank step 20 : conflict free)
#define VT_STR 72    // 64 + 8    (36 dwords  -> lane bank step 36 : conflict free)
#define P_STR  72

// LDS layout offsets in _Float16 units
#define OFF_XS 0
#define OFF_P  0                                  // P overlays XS (XS dead after QKV)
#define SZ_UNION 36864                            // max(64*264=16896, 8*64*72=36864)
#define OFF_Q  SZ_UNION                           // 8*64*40 = 20480  (reused as attn-out)
#define OFF_K  (OFF_Q + 20480)
#define OFF_VT (OFF_K + 20480)                    // 8*32*72 = 18432
#define LDS_F16 (OFF_VT + 18432)                  // 96256 f16 = 192512 bytes

// ---- WMMA fragment loaders -------------------------------------------------
// A 16x32 f16: lane holds row (lane&15); K = kb+{0..15}, kb = lane>=16 ? 16 : 0
__device__ __forceinline__ v16h load_afrag(const _Float16* base, int row0, int stride,
                                           int koff, int lane) {
  const _Float16* p = base + (row0 + (lane & 15)) * stride + koff + ((lane & 16) ? 16 : 0);
  union { uint4 u[2]; v16h v; } t;
  t.u[0] = *(const uint4*)(p);
  t.u[1] = *(const uint4*)(p + 8);
  return t.v;
}

// B 32x16 f16 from LDS (stored as [col][k] contiguous): lane holds col (lane&15);
// K = kb2+{0..15}, kb2 = lane>=16 ? 16 : 0
__device__ __forceinline__ v16h load_bfrag(const _Float16* base, int col0, int stride,
                                           int koff, int lane) {
  const _Float16* p = base + (col0 + (lane & 15)) * stride + koff + ((lane & 16) ? 16 : 0);
  union { uint4 u[2]; v16h v; } t;
  t.u[0] = *(const uint4*)(p);
  t.u[1] = *(const uint4*)(p + 8);
  return t.v;
}

// B 32x16 from a row-major f32 weight matrix in global memory, converted to f16
__device__ __forceinline__ v16h load_bfrag_g(const float* __restrict__ w, int ldw,
                                             int k0, int col0, int lane) {
  const int c  = col0 + (lane & 15);
  const int kb = (lane & 16) ? 16 : 0;
  const float* p = w + (size_t)(k0 + kb) * (size_t)ldw + c;
  v16h v;
#pragma unroll
  for (int e = 0; e < 16; ++e) v[e] = (_Float16)p[(size_t)e * ldw];
  return v;
}

__device__ __forceinline__ v8f wmma16(v16h a, v16h b, v8f c) {
  return __builtin_amdgcn_wmma_f32_16x16x32_f16(false, a, false, b, (short)0, c, false, false);
}

// ---------------------------------------------------------------------------
__global__ __launch_bounds__(256)
void swin_window_attn_kernel(const float* __restrict__ x,        // [4096,64,256]
                             const float* __restrict__ mask,     // [64,64,64]
                             const float* __restrict__ qkv_w,    // [256,768]
                             const float* __restrict__ qkv_b,    // [768]
                             const float* __restrict__ proj_w,   // [256,256]
                             const float* __restrict__ proj_b,   // [256]
                             const float* __restrict__ rpb,      // [225,8]
                             float* __restrict__ out) {          // [4096,64,256]
  extern __shared__ _Float16 S[];
  const int b      = blockIdx.x;
  const int tid    = threadIdx.x;
  const int wv     = tid >> 5;
  const int lane   = tid & 31;
  const int col    = lane & 15;
  const int rowoff = (lane & 16) ? 8 : 0;

  // ---------------- Phase 0: stage x -> LDS as f16 ----------------
  {
    const float* xb = x + (size_t)b * (N_TOK * DIM);
#pragma unroll 4
    for (int i = 0; i < 64; ++i)            // row i, col tid (coalesced 1KB rows)
      S[OFF_XS + i * XS_STR + tid] = (_Float16)xb[i * DIM + tid];
  }
  __syncthreads();

  // ---------------- Phase 1: QKV GEMM [64,256]x[256,768] ----------------
  // 48 N-tiles of 16; wave handles 6. K = 256 -> 8 chunks of 32.
#pragma unroll 1
  for (int t6 = 0; t6 < 6; ++t6) {
    const int nt = wv * 6 + t6;
    const int nb = nt * 16;
    const int which = nb >> 8;              // 0=q 1=k 2=v
    const int head  = (nb >> 5) & 7;
    const int dh    = (nb >> 4) & 1;
    v8f acc[4] = {v8f{}, v8f{}, v8f{}, v8f{}};
#pragma unroll 1
    for (int kc = 0; kc < 8; ++kc) {
      const v16h bW = load_bfrag_g(qkv_w, 3 * DIM, kc * 32, nb, lane);
#pragma unroll
      for (int mt = 0; mt < 4; ++mt) {
        const v16h aX = load_afrag(S + OFF_XS, mt * 16, XS_STR, kc * 32, lane);
        acc[mt] = wmma16(aX, bW, acc[mt]);
      }
    }
    const float cb = qkv_b[nb + col];
    if (which == 0) {
      _Float16* qh = S + OFF_Q + head * (N_TOK * QK_STR);
#pragma unroll
      for (int mt = 0; mt < 4; ++mt)
#pragma unroll
        for (int r = 0; r < 8; ++r)
          qh[(mt * 16 + r + rowoff) * QK_STR + dh * 16 + col] =
              (_Float16)((acc[mt][r] + cb) * SCALE);
    } else if (which == 1) {
      _Float16* kh = S + OFF_K + head * (N_TOK * QK_STR);
#pragma unroll
      for (int mt = 0; mt < 4; ++mt)
#pragma unroll
        for (int r = 0; r < 8; ++r)
          kh[(mt * 16 + r + rowoff) * QK_STR + dh * 16 + col] =
              (_Float16)(acc[mt][r] + cb);
    } else {
      _Float16* vh = S + OFF_VT + head * (HD * VT_STR);   // transposed [d][t]
#pragma unroll
      for (int mt = 0; mt < 4; ++mt)
#pragma unroll
        for (int r = 0; r < 8; ++r)
          vh[(dh * 16 + col) * VT_STR + (mt * 16 + r + rowoff)] =
              (_Float16)(acc[mt][r] + cb);
    }
  }
  __syncthreads();

  // ---------------- Phase 2: per-head attention (wave wv -> head wv) ----------------
  {
    const int h = wv;
    const _Float16* qh  = S + OFF_Q  + h * (N_TOK * QK_STR);
    const _Float16* kh  = S + OFF_K  + h * (N_TOK * QK_STR);
    const _Float16* vth = S + OFF_VT + h * (HD * VT_STR);
    _Float16*       ph  = S + OFF_P  + h * (N_TOK * P_STR);
    const float*  maskp = mask + (size_t)(b & 63) * (N_TOK * N_TOK);

    v16h aQ[4], bK[4];
#pragma unroll
    for (int t = 0; t < 4; ++t) aQ[t] = load_afrag(qh, 16 * t, QK_STR, 0, lane);
#pragma unroll
    for (int t = 0; t < 4; ++t) bK[t] = load_bfrag(kh, 16 * t, QK_STR, 0, lane);

#pragma unroll 1
    for (int mt = 0; mt < 4; ++mt) {
      v8f acc[4] = {v8f{}, v8f{}, v8f{}, v8f{}};
#pragma unroll
      for (int nt = 0; nt < 4; ++nt) acc[nt] = wmma16(aQ[mt], bK[nt], acc[nt]);

      // bias + mask + softmax, one row per (r, lane-half)
#pragma unroll 1
      for (int r = 0; r < 8; ++r) {
        const int i  = mt * 16 + r + rowoff;
        const int xi = i >> 3, yi = i & 7;
        float v[4];
#pragma unroll
        for (int nt = 0; nt < 4; ++nt) {
          const int j  = nt * 16 + col;
          const int xj = j >> 3, yj = j & 7;
          const int idx = (xi - xj + 7) * 15 + (yi - yj + 7);
          v[nt] = acc[nt][r] + rpb[idx * HEADS + h] + maskp[i * N_TOK + j];
        }
        float m = fmaxf(fmaxf(v[0], v[1]), fmaxf(v[2], v[3]));
#pragma unroll
        for (int off = 1; off < 16; off <<= 1) m = fmaxf(m, __shfl_xor(m, off, 32));
        float e[4], s = 0.f;
#pragma unroll
        for (int nt = 0; nt < 4; ++nt) { e[nt] = __expf(v[nt] - m); s += e[nt]; }
#pragma unroll
        for (int off = 1; off < 16; off <<= 1) s += __shfl_xor(s, off, 32);
        const float inv = __frcp_rn(s);
#pragma unroll
        for (int nt = 0; nt < 4; ++nt)
          ph[i * P_STR + nt * 16 + col] = (_Float16)(e[nt] * inv);
      }
    }

    // P[64,64] @ v[64,32] -> O[64,32]; write O over q buffer (proj layout)
    v16h bV[2][2];
#pragma unroll
    for (int nt = 0; nt < 2; ++nt)
#pragma unroll
      for (int kc = 0; kc < 2; ++kc)
        bV[nt][kc] = load_bfrag(vth, nt * 16, VT_STR, kc * 32, lane);

    _Float16* oh = S + OFF_Q + h * (N_TOK * QK_STR);
#pragma unroll 1
    for (int mt = 0; mt < 4; ++mt) {
      v8f o[2] = {v8f{}, v8f{}};
#pragma unroll
      for (int kc = 0; kc < 2; ++kc) {
        const v16h aP = load_afrag(ph, mt * 16, P_STR, kc * 32, lane);
        o[0] = wmma16(aP, bV[0][kc], o[0]);
        o[1] = wmma16(aP, bV[1][kc], o[1]);
      }
#pragma unroll
      for (int nt = 0; nt < 2; ++nt)
#pragma unroll
        for (int r = 0; r < 8; ++r)
          oh[(mt * 16 + r + rowoff) * QK_STR + nt * 16 + col] = (_Float16)o[nt][r];
    }
  }
  __syncthreads();

  // ---------------- Phase 3: proj GEMM [64,256]x[256,256] + bias ----------------
  {
    v8f acc[4][2] = {{v8f{}, v8f{}}, {v8f{}, v8f{}}, {v8f{}, v8f{}}, {v8f{}, v8f{}}};
#pragma unroll 1
    for (int kc = 0; kc < 8; ++kc) {
      const _Float16* ob = S + OFF_Q + kc * (N_TOK * QK_STR);  // channels kc*32..+31
      v16h aO[4];
#pragma unroll
      for (int mt = 0; mt < 4; ++mt) aO[mt] = load_afrag(ob, mt * 16, QK_STR, 0, lane);
      v16h bW[2];
#pragma unroll
      for (int nt = 0; nt < 2; ++nt)
        bW[nt] = load_bfrag_g(proj_w, DIM, kc * 32, wv * 32 + nt * 16, lane);
#pragma unroll
      for (int mt = 0; mt < 4; ++mt) {
        acc[mt][0] = wmma16(aO[mt], bW[0], acc[mt][0]);
        acc[mt][1] = wmma16(aO[mt], bW[1], acc[mt][1]);
      }
    }
    float* ob = out + (size_t)b * (N_TOK * DIM);
#pragma unroll
    for (int nt = 0; nt < 2; ++nt) {
      const int j  = wv * 32 + nt * 16 + col;
      const float pb = proj_b[j];
#pragma unroll
      for (int mt = 0; mt < 4; ++mt)
#pragma unroll
        for (int r = 0; r < 8; ++r)
          ob[(mt * 16 + r + rowoff) * DIM + j] = acc[mt][nt][r] + pb;
    }
  }
}

extern "C" void kernel_launch(void* const* d_in, const int* in_sizes, int n_in,
                              void* d_out, int out_size, void* d_ws, size_t ws_size,
                              hipStream_t stream) {
  (void)in_sizes; (void)n_in; (void)out_size; (void)d_ws; (void)ws_size;
  const float* x      = (const float*)d_in[0];
  const float* mask   = (const float*)d_in[1];
  const float* qkv_w  = (const float*)d_in[2];
  const float* qkv_b  = (const float*)d_in[3];
  const float* proj_w = (const float*)d_in[4];
  const float* proj_b = (const float*)d_in[5];
  const float* rpb    = (const float*)d_in[6];
  float* out = (float*)d_out;

  const size_t lds_bytes = (size_t)LDS_F16 * sizeof(_Float16);   // 192512 B
  hipError_t err = hipFuncSetAttribute((const void*)swin_window_attn_kernel,
                                       hipFuncAttributeMaxDynamicSharedMemorySize,
                                       (int)lds_bytes);
  (void)err;
  swin_window_attn_kernel<<<4096, 256, lds_bytes, stream>>>(
      x, mask, qkv_w, qkv_b, proj_w, proj_b, rpb, out);
}
